// MultiHeadAttention_46239617909404
// MI455X (gfx1250) — matrix operand order repaired and checked by reading, not compiled, not run on hardware
//
#include <hip/hip_runtime.h>

// ---------------------------------------------------------------------------
// MHA layer for MI455X (gfx1250, wave32, WMMA 16x16x32 bf16,
// async global->LDS staging with ASYNCcnt-based double buffering)
// B=4, T=2048, C=1024, H=16, D=64
// ---------------------------------------------------------------------------

#define B_  4
#define T_  2048
#define C_  1024
#define H_  16
#define D_  64
#define C3_ (3 * C_)

typedef __attribute__((ext_vector_type(16))) __bf16 v16bf;
typedef __attribute__((ext_vector_type(8)))  float  v8f;
typedef __attribute__((ext_vector_type(4)))  int    v4i;

union Frag {
  unsigned u[8];
  v16bf    v;
};
static_assert(sizeof(Frag) == 32, "frag size");

// ---------------------------------------------------------------------------
// Async global->LDS copy (CDNA5 GLOBAL_LOAD_ASYNC_TO_LDS_B128, ASYNCcnt).
// ---------------------------------------------------------------------------
#if defined(__has_builtin)
#if __has_builtin(__builtin_amdgcn_global_load_async_to_lds_b128)
#define HAVE_ASYNC_LDS 1
#endif
#if __has_builtin(__builtin_amdgcn_s_wait_asynccnt)
#define HAVE_WAIT_ASYNC 1
#endif
#endif

typedef __attribute__((address_space(1))) v4i gv4i;   // global int32x4
typedef __attribute__((address_space(3))) v4i lv4i;   // LDS int32x4

__device__ __forceinline__ void async_b128(void* lds_dst, const void* gsrc) {
#if defined(HAVE_ASYNC_LDS)
  __builtin_amdgcn_global_load_async_to_lds_b128(
      (gv4i*)(void*)gsrc, (lv4i*)lds_dst, 0, 0);
#else
  *(uint4*)lds_dst = *(const uint4*)gsrc;
#endif
}

__device__ __forceinline__ void wait_async0() {
#if defined(HAVE_ASYNC_LDS)
#if defined(HAVE_WAIT_ASYNC)
  __builtin_amdgcn_s_wait_asynccnt(0);
#else
  asm volatile("s_wait_asynccnt 0" ::: "memory");
#endif
#endif
}

__device__ __forceinline__ unsigned short f32_to_bf16(float f) {
  unsigned u = __float_as_uint(f);
  unsigned r = 0x7FFFu + ((u >> 16) & 1u);       // round-to-nearest-even
  return (unsigned short)((u + r) >> 16);
}

// ISA 16-bit A-matrix 16x32 layout: lane half hl picks K-phase, VGPR v covers
// K pairs: v<4 -> 8h+2v, v>=4 -> 16+8h+2(v-4).
__device__ __forceinline__ int koffA(int v, int hl) {
  return (v < 4) ? (8 * hl + 2 * v) : (16 + 8 * hl + 2 * (v - 4));
}
// 16-bit B-matrix 32x16 layout: same K order as A: v<4 -> 8h+2v, v>=4 -> 16+8h+2(v-4).
__device__ __forceinline__ int koffB(int v, int hl) {
  return (v < 4) ? (8 * hl + 2 * v) : (16 + 8 * hl + 2 * (v - 4));
}

// ---------------------------------------------------------------------------
// fp32 -> bf16 conversion (vectorized x4)
// ---------------------------------------------------------------------------
__global__ __launch_bounds__(256) void cvt_f32_bf16(const float* __restrict__ in,
                                                    unsigned short* __restrict__ out,
                                                    int n4) {
  int i = blockIdx.x * blockDim.x + threadIdx.x;
  if (i >= n4) return;
  int idx = i * 4;
  float4 f = *(const float4*)(in + idx);
  uint2 r;
  r.x = (unsigned)f32_to_bf16(f.x) | ((unsigned)f32_to_bf16(f.y) << 16);
  r.y = (unsigned)f32_to_bf16(f.z) | ((unsigned)f32_to_bf16(f.w) << 16);
  *(uint2*)(out + idx) = r;
}

// ---------------------------------------------------------------------------
// bf16 GEMM: C(MxN) = A(MxK) * B(KxN), WMMA f32_16x16x32_bf16
// Block: 256 threads (8 waves), BM=128 BN=64 BK=32, wave tile 32x32.
// Double-buffered LDS tiles; A staged via async-to-LDS for overlap.
// ---------------------------------------------------------------------------
#define BM 128
#define BN 64
#define BK 32
#define ASTR (BK + 8)   // 40 ushorts = 80B rows (16B aligned)

template <bool OUT_BF16>
__global__ __launch_bounds__(256) void gemm_bf16_wmma(
    const unsigned short* __restrict__ A,
    const unsigned short* __restrict__ B,
    void* __restrict__ Cout,
    int M, int N, int K) {
  __shared__ unsigned short Ash[2][BM][ASTR];
  __shared__ unsigned short Bsh[2][BN][ASTR];   // stored transposed: [n][k]

  const int tid  = threadIdx.x;
  const int wave = tid >> 5;
  const int lane = tid & 31;
  const int l16  = lane & 15;
  const int hl   = lane >> 4;
  const int wRow = wave >> 1;   // 0..3
  const int wCol = wave & 1;    // 0..1
  const int blockM = blockIdx.y * BM;
  const int blockN = blockIdx.x * BN;

  const v8f zf = {0.f, 0.f, 0.f, 0.f, 0.f, 0.f, 0.f, 0.f};
  v8f acc[2][2];
  acc[0][0] = zf; acc[0][1] = zf; acc[1][0] = zf; acc[1][1] = zf;

  auto stage = [&](int k0, int p) {
    // A tile 128x32: 512 8-elem chunks, 2 per thread, async to LDS.
#pragma unroll
    for (int c = 0; c < 2; ++c) {
      int chunk = c * 256 + tid;
      int r = chunk >> 2;
      int cc = (chunk & 3) * 8;
      async_b128(&Ash[p][r][cc], A + (size_t)(blockM + r) * K + k0 + cc);
    }
    // B tile 32x64 -> transposed into Bsh[p][n][k] (manual scatter).
    {
      int kr = tid >> 3;            // 0..31
      int nc = (tid & 7) * 8;       // 0..56
      uint4 val = *(const uint4*)(B + (size_t)(k0 + kr) * N + blockN + nc);
      const unsigned short* e = (const unsigned short*)&val;
#pragma unroll
      for (int j = 0; j < 8; ++j) Bsh[p][nc + j][kr] = e[j];
    }
  };

  stage(0, 0);
  int p = 0;
  for (int k0 = 0; k0 < K; k0 += BK, p ^= 1) {
    wait_async0();          // this wave's async writes to tile p complete
    __syncthreads();        // all waves' writes visible
    if (k0 + BK < K) stage(k0 + BK, p ^ 1);   // overlap next tile with compute

    Frag af[2], bf[2];
#pragma unroll
    for (int tm = 0; tm < 2; ++tm) {
      int m = wRow * 32 + tm * 16 + l16;
#pragma unroll
      for (int v = 0; v < 8; ++v)
        af[tm].u[v] = *(const unsigned*)(&Ash[p][m][koffA(v, hl)]);
    }
#pragma unroll
    for (int tn = 0; tn < 2; ++tn) {
      int n = wCol * 32 + tn * 16 + l16;
#pragma unroll
      for (int v = 0; v < 8; ++v)
        bf[tn].u[v] = *(const unsigned*)(&Bsh[p][n][koffB(v, hl)]);
    }
#pragma unroll
    for (int tm = 0; tm < 2; ++tm)
#pragma unroll
      for (int tn = 0; tn < 2; ++tn)
        acc[tm][tn] = __builtin_amdgcn_wmma_f32_16x16x32_bf16(
            false, af[tm].v, false, bf[tn].v, (short)0, acc[tm][tn], false, false);

    __syncthreads();        // reads of tile p done before it is re-staged
  }

  // C/D layout: lane = N col (mod 16), VGPR v = row v + 8*half
#pragma unroll
  for (int tm = 0; tm < 2; ++tm) {
#pragma unroll
    for (int tn = 0; tn < 2; ++tn) {
      const int n  = blockN + wCol * 32 + tn * 16 + l16;
      const int mB = blockM + wRow * 32 + tm * 16 + 8 * hl;
#pragma unroll
      for (int v = 0; v < 8; ++v) {
        float val = acc[tm][tn][v];
        if (OUT_BF16)
          ((unsigned short*)Cout)[(size_t)(mB + v) * N + n] = f32_to_bf16(val);
        else
          ((float*)Cout)[(size_t)(mB + v) * N + n] = val;
      }
    }
  }
}

// ---------------------------------------------------------------------------
// Flash attention (causal), WMMA bf16. 128 threads = 4 waves; each wave owns
// 16 query rows; block owns 64 queries of one (b,h). Key blocks of 64.
// K tile staged async to LDS; V staged transposed (manual scatter).
// All WMMA fragments preloaded into distinct registers so the 8-WMMA groups
// issue back-to-back after a single LDS wait.
// ---------------------------------------------------------------------------
#define QB 64
#define KB 64
#define KSTR (D_ + 8)   // 72 ushorts = 144B (16B aligned)
#define VSTR (KB + 8)

__global__ __launch_bounds__(128) void flash_attn_wmma(
    const unsigned short* __restrict__ qkv,
    unsigned short* __restrict__ y) {
  __shared__ unsigned short Ksh[KB][KSTR];      // [key][d]
  __shared__ unsigned short Vt[D_][VSTR];       // [d][key] (transposed)
  __shared__ unsigned short Psh[4][16][VSTR];   // per-wave P staging

  const int bh = blockIdx.x;
  const int b  = bh / H_;
  const int h  = bh % H_;
  const int qb = blockIdx.y;
  const int qBase = qb * QB;

  const int tid  = threadIdx.x;
  const int wave = tid >> 5;
  const int lane = tid & 31;
  const int l16  = lane & 15;
  const int hl   = lane >> 4;

  const size_t baseQ = (size_t)b * T_ * C3_ + (size_t)h * D_;
  const size_t baseK = baseQ + C_;
  const size_t baseV = baseQ + 2 * C_;

  // Q fragments for this wave's 16 rows (A-matrix layout, contraction = d)
  Frag qf[2];
  {
    const int qrow = qBase + wave * 16 + l16;
    const unsigned short* qp = qkv + baseQ + (size_t)qrow * C3_;
#pragma unroll
    for (int ds = 0; ds < 2; ++ds)
#pragma unroll
      for (int v = 0; v < 8; ++v)
        qf[ds].u[v] = *(const unsigned*)(qp + ds * 32 + koffA(v, hl));
  }

  const v8f zf = {0.f, 0.f, 0.f, 0.f, 0.f, 0.f, 0.f, 0.f};
  v8f o[4];
  o[0] = zf; o[1] = zf; o[2] = zf; o[3] = zf;
  float m_i[8], l_i[8];
#pragma unroll
  for (int v = 0; v < 8; ++v) { m_i[v] = -1.0e30f; l_i[v] = 0.f; }

  for (int j = 0; j <= qb; ++j) {
    const int kBase = j * KB;
    // --- K block (64x64) async row-major; V block transposed (manual) ---
#pragma unroll
    for (int c = 0; c < 4; ++c) {
      int chunk = c * 128 + tid;     // 0..511
      int r  = chunk >> 3;
      int dc = (chunk & 7) * 8;
      async_b128(&Ksh[r][dc], qkv + baseK + (size_t)(kBase + r) * C3_ + dc);
      uint4 vval = *(const uint4*)(qkv + baseV + (size_t)(kBase + r) * C3_ + dc);
      const unsigned short* e = (const unsigned short*)&vval;
#pragma unroll
      for (int jj = 0; jj < 8; ++jj) Vt[dc + jj][r] = e[jj];
    }
    wait_async0();
    __syncthreads();

    // --- S = Q K^T : preload all 8 B-fragments, then 8 WMMAs back-to-back ---
    Frag kbf[4][2];
#pragma unroll
    for (int kt = 0; kt < 4; ++kt) {
      const int key = kt * 16 + l16;
#pragma unroll
      for (int ds = 0; ds < 2; ++ds)
#pragma unroll
        for (int v = 0; v < 8; ++v)
          kbf[kt][ds].u[v] = *(const unsigned*)(&Ksh[key][ds * 32 + koffB(v, hl)]);
    }
    v8f s[4];
    s[0] = zf; s[1] = zf; s[2] = zf; s[3] = zf;
#pragma unroll
    for (int kt = 0; kt < 4; ++kt)
#pragma unroll
      for (int ds = 0; ds < 2; ++ds)
        s[kt] = __builtin_amdgcn_wmma_f32_16x16x32_bf16(
            false, qf[ds].v, false, kbf[kt][ds].v, (short)0, s[kt], false, false);

    // --- scale, causal mask, online softmax ---
    float p[4][8];
    float rowmax[8];
#pragma unroll
    for (int v = 0; v < 8; ++v) rowmax[v] = -1.0e30f;
#pragma unroll
    for (int kt = 0; kt < 4; ++kt) {
      const int key = kBase + kt * 16 + l16;
#pragma unroll
      for (int v = 0; v < 8; ++v) {
        const int qr = qBase + wave * 16 + v + 8 * hl;
        float val = s[kt][v] * 0.125f;                 // 1/sqrt(64)
        if (key > qr) val = -1.0e30f;
        p[kt][v] = val;
        rowmax[v] = fmaxf(rowmax[v], val);
      }
    }
#pragma unroll
    for (int v = 0; v < 8; ++v)
#pragma unroll
      for (int mk = 1; mk < 16; mk <<= 1)
        rowmax[v] = fmaxf(rowmax[v], __shfl_xor(rowmax[v], mk, 32));

    float alpha[8];
#pragma unroll
    for (int v = 0; v < 8; ++v) {
      float mnew = fmaxf(m_i[v], rowmax[v]);
      alpha[v] = __expf(m_i[v] - mnew);
      m_i[v] = mnew;
    }
    float rs[8];
#pragma unroll
    for (int v = 0; v < 8; ++v) rs[v] = 0.f;
#pragma unroll
    for (int kt = 0; kt < 4; ++kt)
#pragma unroll
      for (int v = 0; v < 8; ++v) {
        float e = __expf(p[kt][v] - m_i[v]);
        p[kt][v] = e;
        rs[v] += e;
      }
#pragma unroll
    for (int v = 0; v < 8; ++v)
#pragma unroll
      for (int mk = 1; mk < 16; mk <<= 1)
        rs[v] += __shfl_xor(rs[v], mk, 32);
#pragma unroll
    for (int v = 0; v < 8; ++v) l_i[v] = l_i[v] * alpha[v] + rs[v];
#pragma unroll
    for (int dt = 0; dt < 4; ++dt)
#pragma unroll
      for (int v = 0; v < 8; ++v) o[dt][v] = o[dt][v] * alpha[v];

    // --- re-shape P (C-layout) -> A-layout via LDS (per-wave, in-order DS) ---
#pragma unroll
    for (int kt = 0; kt < 4; ++kt)
#pragma unroll
      for (int v = 0; v < 8; ++v)
        Psh[wave][v + 8 * hl][kt * 16 + l16] = f32_to_bf16(p[kt][v]);

    // --- O += P V : preload all fragments, then 8 WMMAs back-to-back ---
    Frag pf[2];
    Frag vfr[2][4];
#pragma unroll
    for (int ks = 0; ks < 2; ++ks) {
#pragma unroll
      for (int v = 0; v < 8; ++v)
        pf[ks].u[v] = *(const unsigned*)(&Psh[wave][l16][ks * 32 + koffA(v, hl)]);
#pragma unroll
      for (int dt = 0; dt < 4; ++dt) {
        const int dcol = dt * 16 + l16;
#pragma unroll
        for (int v = 0; v < 8; ++v)
          vfr[ks][dt].u[v] = *(const unsigned*)(&Vt[dcol][ks * 32 + koffB(v, hl)]);
      }
    }
#pragma unroll
    for (int ks = 0; ks < 2; ++ks)
#pragma unroll
      for (int dt = 0; dt < 4; ++dt)
        o[dt] = __builtin_amdgcn_wmma_f32_16x16x32_bf16(
            false, pf[ks].v, false, vfr[ks][dt].v, (short)0, o[dt], false, false);

    __syncthreads();
  }

  // --- normalize and store y (bf16, (B,T,C) with col = h*64 + d) ---
  float inv[8];
#pragma unroll
  for (int v = 0; v < 8; ++v) inv[v] = 1.0f / l_i[v];
#pragma unroll
  for (int dt = 0; dt < 4; ++dt) {
    const int dcol = dt * 16 + l16;
#pragma unroll
    for (int v = 0; v < 8; ++v) {
      const int qr = qBase + wave * 16 + v + 8 * hl;
      y[(size_t)(b * T_ + qr) * C_ + h * D_ + dcol] = f32_to_bf16(o[dt][v] * inv[v]);
    }
  }
}

// ---------------------------------------------------------------------------
// Host launch
// ---------------------------------------------------------------------------
extern "C" void kernel_launch(void* const* d_in, const int* in_sizes, int n_in,
                              void* d_out, int out_size, void* d_ws, size_t ws_size,
                              hipStream_t stream) {
  const float* x     = (const float*)d_in[0];   // (B,T,C)
  const float* Wqkv  = (const float*)d_in[1];   // (C,3C)
  const float* Wproj = (const float*)d_in[2];   // (C,C)
  float* out = (float*)d_out;                   // (B,T,C)

  char* ws = (char*)d_ws;
  const size_t MiB = 1024 * 1024;
  unsigned short* xb    = (unsigned short*)(ws + 0);          // 16 MiB
  unsigned short* wqkvb = (unsigned short*)(ws + 16 * MiB);   // 6 MiB
  unsigned short* wprjb = (unsigned short*)(ws + 22 * MiB);   // 2 MiB
  unsigned short* qkvb  = (unsigned short*)(ws + 24 * MiB);   // 48 MiB
  unsigned short* yb    = (unsigned short*)(ws + 72 * MiB);   // 16 MiB

  const int nX  = B_ * T_ * C_;        // 8388608
  const int nWq = C_ * C3_;            // 3145728
  const int nWp = C_ * C_;             // 1048576

  cvt_f32_bf16<<<(nX / 4 + 255) / 256, 256, 0, stream>>>(x, xb, nX / 4);
  cvt_f32_bf16<<<(nWq / 4 + 255) / 256, 256, 0, stream>>>(Wqkv, wqkvb, nWq / 4);
  cvt_f32_bf16<<<(nWp / 4 + 255) / 256, 256, 0, stream>>>(Wproj, wprjb, nWp / 4);

  // qkv = x @ Wqkv : (8192 x 1024) * (1024 x 3072) -> bf16
  {
    dim3 grid(C3_ / BN, (B_ * T_) / BM);
    gemm_bf16_wmma<true><<<grid, 256, 0, stream>>>(xb, wqkvb, qkvb,
                                                   B_ * T_, C3_, C_);
  }

  // attention: grid (B*H, T/QB)
  {
    dim3 grid(B_ * H_, T_ / QB);
    flash_attn_wmma<<<grid, 128, 0, stream>>>(qkvb, yb);
  }

  // out = y @ Wproj : (8192 x 1024) * (1024 x 1024) -> fp32
  {
    dim3 grid(C_ / BN, (B_ * T_) / BM);
    gemm_bf16_wmma<false><<<grid, 256, 0, stream>>>(yb, wprjb, out,
                                                    B_ * T_, C_, C_);
  }
}
